// CHARM_89146341196444
// MI455X (gfx1250) — matrix operand order repaired and checked by reading, not compiled, not run on hardware
//
#include <hip/hip_runtime.h>
#include <hip/hip_bf16.h>

#define FEAT 1024
#define HID  512
#define DK   256
#define NROW 8192

// padded LDS row stride (halves): 80B rows -> bank-conflict-free fragments,
// still 16B-aligned everywhere (80 = 5*16).
#define LSTR 40

typedef __attribute__((ext_vector_type(16))) _Float16 v16h;
typedef __attribute__((ext_vector_type(8)))  float    v8f;

union Frag { v16h v; uint4 u[2]; };
union F4   { float4 v; float f[4]; };
union Pk4  { uint2 u; _Float16 h[4]; };

// ---------------------------------------------------------------------------
// Kernel 0: f32 -> f16 convert (X -> Xh)
// ---------------------------------------------------------------------------
__global__ __launch_bounds__(256) void f32_to_f16(const float* __restrict__ x,
                                                  _Float16* __restrict__ y, int n4) {
  int i = blockIdx.x * blockDim.x + threadIdx.x;
  if (i < n4) {
    float4 f = ((const float4*)x)[i];
    _Float16* o = y + (size_t)i * 4;
    o[0] = (_Float16)f.x; o[1] = (_Float16)f.y;
    o[2] = (_Float16)f.z; o[3] = (_Float16)f.w;
  }
}

// ---------------------------------------------------------------------------
// Tiled WMMA GEMM: C[M x N] = act(A[M x K](f16) * B[K x N](f32->f16) + bias)
// Block tile 64x128, 8 waves (4 row-tiles x 2 col-groups), wave tile 16x64,
// K-step 32 -> v_wmma_f32_16x16x32_f16.
// A tile staged via gfx1250 global_load_async_to_lds_b128 (ASYNCcnt).
// B tile staged transposed [n][k] with packed ds_store_b64.
// ---------------------------------------------------------------------------
template <bool OUT_F16, bool RELU>
__global__ __launch_bounds__(256) void gemm_wmma(
    const _Float16* __restrict__ A, int lda,
    const float* __restrict__ B, int ldb,
    const float* __restrict__ bias,
    void* __restrict__ Cout, int ldc, int K) {
  __shared__ __align__(16) _Float16 Als[64 * LSTR];
  __shared__ __align__(16) _Float16 Bls[128 * LSTR];

  const int t = threadIdx.x;
  const int w = t >> 5, lane = t & 31;
  const int rowTile = w & 3, colGrp = w >> 2;
  const int mn = lane & 15, half = lane >> 4;

  v8f acc[4] = {};

  // --- staging roles ---
  // A: 64 rows x 32 halves; thread -> one b128 (8 halves)
  const int arow = t >> 2, apart = t & 3;
  const _Float16* aSrc = A + (size_t)(blockIdx.x * 64 + arow) * lda + apart * 8;
  const unsigned aLds =
      (unsigned)(size_t)(void*)(Als + arow * LSTR + apart * 8);
  // B: 32 k-rows x 128 cols; thread -> 4k x 4n micro-tile.
  // kq = wave id keeps the four b128 row loads fully coalesced (512B/row/wave).
  const int kq = t >> 5, ng = t & 31;
  const float* bSrc = B + (size_t)(kq * 4) * ldb + blockIdx.y * 128 + ng * 4;

  for (int k0 = 0; k0 < K; k0 += 32) {
    __syncthreads();  // previous tile fully consumed

    // ---- stage A (async DMA into LDS, no VGPR round trip) ----
    {
      const _Float16* ga = aSrc + k0;
      asm volatile("global_load_async_to_lds_b128 %0, %1, off"
                   :: "v"(aLds), "v"(ga)
                   : "memory");
    }

    // ---- stage B transposed: Bls[n][k] as f16, packed b64 stores ----
    const float* bs = bSrc + (size_t)k0 * ldb;
    F4 ld[4];
#pragma unroll
    for (int r = 0; r < 4; ++r)
      ld[r].v = *(const float4*)(bs + (size_t)r * ldb);
#pragma unroll
    for (int c = 0; c < 4; ++c) {
      Pk4 pk;
      pk.h[0] = (_Float16)ld[0].f[c];
      pk.h[1] = (_Float16)ld[1].f[c];
      pk.h[2] = (_Float16)ld[2].f[c];
      pk.h[3] = (_Float16)ld[3].f[c];
      *(uint2*)(Bls + (ng * 4 + c) * LSTR + kq * 4) = pk.u;
    }
    if (k0 + 32 < K)  // gfx1250 global_prefetch_b8 of next B tile
      __builtin_prefetch(bs + (size_t)32 * ldb, 0, 0);

    asm volatile("s_wait_asynccnt 0" ::: "memory");  // our async A chunk landed
    __syncthreads();                                  // everyone's tile visible

    // ---- compute ----
    // A fragment: lane holds row mn; k = half*8..+7 and half*8+16..+23
    Frag a;
    const uint4* ap = (const uint4*)(Als + (rowTile * 16 + mn) * LSTR + half * 8);
    a.u[0] = ap[0];
    a.u[1] = ap[2];  // +16 halves = +32B
#pragma unroll
    for (int sub = 0; sub < 4; ++sub) {
      // B fragment: lane holds column mn; k = half*8..+7 and half*8+16..+23
      Frag b;
      const uint4* bp =
          (const uint4*)(Bls + (colGrp * 64 + sub * 16 + mn) * LSTR + half * 8);
      b.u[0] = bp[0];
      b.u[1] = bp[2];
      acc[sub] = __builtin_amdgcn_wmma_f32_16x16x32_f16(
          false, a.v, false, b.v, (short)0, acc[sub], false, false);
    }
  }

  // Epilogue: C/D layout -> VGPR r holds M=r (lanes 0-15) / M=r+8 (16-31), N=lane&15
#pragma unroll
  for (int sub = 0; sub < 4; ++sub) {
    int col = blockIdx.y * 128 + colGrp * 64 + sub * 16 + mn;
    float bb = bias ? bias[col] : 0.0f;
#pragma unroll
    for (int r = 0; r < 8; ++r) {
      int row = blockIdx.x * 64 + rowTile * 16 + r + half * 8;
      float vv = acc[sub][r] + bb;
      if (RELU) vv = fmaxf(vv, 0.0f);
      if (OUT_F16)
        ((_Float16*)Cout)[(size_t)row * ldc + col] = (_Float16)vv;
      else
        ((float*)Cout)[(size_t)row * ldc + col] = vv;
    }
  }
}

// ---------------------------------------------------------------------------
// Kernel 3: sparse-gather attention scores. One wave per row:
//   a_raw_kk[i] = (1/16) * sum_{j<=kk} dot(q[i], k[sp[i,j]])
// ---------------------------------------------------------------------------
__global__ __launch_bounds__(256) void attn_gather(
    const float* __restrict__ q, const float* __restrict__ kmat,
    const int* __restrict__ sp, float* __restrict__ araw) {
  const int lane = threadIdx.x & 31;
  const int row = blockIdx.x * 8 + (threadIdx.x >> 5);
  const float* qp = q + (size_t)row * DK + lane * 8;
  float4 q0 = ((const float4*)qp)[0];
  float4 q1 = ((const float4*)qp)[1];
  float a2 = 0.f, a4 = 0.f, asum = 0.f;
#pragma unroll
  for (int j = 0; j < 9; ++j) {
    int idx = sp[row * 9 + j];
    const float* kp = kmat + (size_t)idx * DK + lane * 8;
    float4 k0 = ((const float4*)kp)[0];
    float4 k1 = ((const float4*)kp)[1];
    float d = q0.x * k0.x + q0.y * k0.y + q0.z * k0.z + q0.w * k0.w +
              q1.x * k1.x + q1.y * k1.y + q1.z * k1.z + q1.w * k1.w;
#pragma unroll
    for (int m = 16; m; m >>= 1) d += __shfl_xor(d, m, 32);
    asum += d;
    if (j == 2) a2 = asum;
    if (j == 4) a4 = asum;
  }
  if (lane == 0) {
    const float sc = 0.0625f;  // 1/sqrt(256)
    araw[row * 3 + 0] = a2 * sc;
    araw[row * 3 + 1] = a4 * sc;
    araw[row * 3 + 2] = asum * sc;
  }
}

// ---------------------------------------------------------------------------
// Kernel 4: three N-wide softmaxes -> combined coefficient per row.
// Also zero-inits s[DK] (ws is poisoned between runs).
// ---------------------------------------------------------------------------
__global__ __launch_bounds__(1024) void softmax_coeff(
    const float* __restrict__ araw, float* __restrict__ coeff,
    float* __restrict__ s) {
  __shared__ float red[1024];
  const int t = threadIdx.x;
  float mloc[3] = {-1e30f, -1e30f, -1e30f};
  for (int i = t; i < NROW; i += 1024) {
    mloc[0] = fmaxf(mloc[0], araw[i * 3 + 0]);
    mloc[1] = fmaxf(mloc[1], araw[i * 3 + 1]);
    mloc[2] = fmaxf(mloc[2], araw[i * 3 + 2]);
  }
  float mx[3];
  for (int c = 0; c < 3; ++c) {
    red[t] = mloc[c];
    __syncthreads();
    for (int off = 512; off; off >>= 1) {
      if (t < off) red[t] = fmaxf(red[t], red[t + off]);
      __syncthreads();
    }
    mx[c] = red[0];
    __syncthreads();
  }
  float zloc[3] = {0.f, 0.f, 0.f};
  for (int i = t; i < NROW; i += 1024) {
    zloc[0] += expf(araw[i * 3 + 0] - mx[0]);
    zloc[1] += expf(araw[i * 3 + 1] - mx[1]);
    zloc[2] += expf(araw[i * 3 + 2] - mx[2]);
  }
  float iz[3];
  for (int c = 0; c < 3; ++c) {
    red[t] = zloc[c];
    __syncthreads();
    for (int off = 512; off; off >>= 1) {
      if (t < off) red[t] += red[t + off];
      __syncthreads();
    }
    iz[c] = 1.0f / red[0];
    __syncthreads();
  }
  for (int i = t; i < NROW; i += 1024) {
    float co = expf(araw[i * 3 + 0] - mx[0]) * iz[0] +
               expf(araw[i * 3 + 1] - mx[1]) * iz[1] +
               expf(araw[i * 3 + 2] - mx[2]) * iz[2];
    coeff[i] = co;
  }
  if (t < DK) s[t] = 0.0f;
}

// ---------------------------------------------------------------------------
// Kernel 5: s[col] += sum_i coeff[i] * value[i, col]
// ---------------------------------------------------------------------------
__global__ __launch_bounds__(256) void weighted_sum(
    const float* __restrict__ coeff, const float* __restrict__ v,
    float* __restrict__ s) {
  const int col = threadIdx.x;
  const int base = blockIdx.x * 128;
  float acc = 0.f;
  for (int r = 0; r < 128; ++r)
    acc += coeff[base + r] * v[(size_t)(base + r) * DK + col];
  atomicAdd(&s[col], acc);
}

// ---------------------------------------------------------------------------
// Kernel 6: logits = s @ Wout + bout ; out = softmax(logits)
// ---------------------------------------------------------------------------
__global__ void final_head(const float* __restrict__ s,
                           const float* __restrict__ Wout,
                           const float* __restrict__ bout,
                           float* __restrict__ out) {
  if (threadIdx.x == 0) {
    float l0 = bout[0], l1 = bout[1];
    for (int c = 0; c < DK; ++c) {
      l0 += s[c] * Wout[c * 2 + 0];
      l1 += s[c] * Wout[c * 2 + 1];
    }
    float m = fmaxf(l0, l1);
    float e0 = expf(l0 - m), e1 = expf(l1 - m);
    float inv = 1.0f / (e0 + e1);
    out[0] = e0 * inv;
    out[1] = e1 * inv;
  }
}

extern "C" void kernel_launch(void* const* d_in, const int* in_sizes, int n_in,
                              void* d_out, int out_size, void* d_ws, size_t ws_size,
                              hipStream_t stream) {
  (void)in_sizes; (void)n_in; (void)out_size; (void)ws_size;
  const float* X    = (const float*)d_in[0];
  const int*   sp   = (const int*)d_in[1];
  const float* W1   = (const float*)d_in[2];
  const float* b1   = (const float*)d_in[3];
  const float* Wq   = (const float*)d_in[4];
  const float* Wk   = (const float*)d_in[5];
  const float* Wv   = (const float*)d_in[6];
  const float* bv   = (const float*)d_in[7];
  const float* Wout = (const float*)d_in[8];
  const float* bout = (const float*)d_in[9];
  float* out = (float*)d_out;

  char* p = (char*)d_ws;
  auto alloc = [&](size_t bytes) -> char* {
    char* r = p;
    p += (bytes + 255) & ~(size_t)255;
    return r;
  };
  _Float16* Xh = (_Float16*)alloc((size_t)NROW * FEAT * 2);  // 16 MB
  _Float16* Dh = (_Float16*)alloc((size_t)NROW * HID * 2);   // 8 MB
  float* qb    = (float*)alloc((size_t)NROW * DK * 4);       // 8 MB
  float* kb    = (float*)alloc((size_t)NROW * DK * 4);       // 8 MB
  float* vb    = (float*)alloc((size_t)NROW * DK * 4);       // 8 MB
  float* araw  = (float*)alloc((size_t)NROW * 3 * 4);
  float* coeff = (float*)alloc((size_t)NROW * 4);
  float* s     = (float*)alloc((size_t)DK * 4);

  f32_to_f16<<<(NROW * FEAT / 4 + 255) / 256, 256, 0, stream>>>(X, Xh, NROW * FEAT / 4);

  // dense = relu(X @ W1 + b1), f16 out
  gemm_wmma<true, true><<<dim3(NROW / 64, HID / 128), 256, 0, stream>>>(
      Xh, FEAT, W1, HID, b1, (void*)Dh, HID, FEAT);
  // q / k / v
  gemm_wmma<false, false><<<dim3(NROW / 64, DK / 128), 256, 0, stream>>>(
      Dh, HID, Wq, DK, nullptr, (void*)qb, DK, HID);
  gemm_wmma<false, false><<<dim3(NROW / 64, DK / 128), 256, 0, stream>>>(
      Dh, HID, Wk, DK, nullptr, (void*)kb, DK, HID);
  gemm_wmma<false, false><<<dim3(NROW / 64, DK / 128), 256, 0, stream>>>(
      Dh, HID, Wv, DK, bv, (void*)vb, DK, HID);

  attn_gather<<<NROW / 8, 256, 0, stream>>>(qb, kb, sp, araw);
  softmax_coeff<<<1, 1024, 0, stream>>>(araw, coeff, s);
  weighted_sum<<<NROW / 128, 256, 0, stream>>>(coeff, vb, s);
  final_head<<<1, 64, 0, stream>>>(s, Wout, bout, out);
}
